// MultiHeadSelfAttention_32212254720318
// MI455X (gfx1250) — matrix operand order repaired and checked by reading, not compiled, not run on hardware
//
#include <hip/hip_runtime.h>
#include <cstdint>

typedef __attribute__((ext_vector_type(16))) _Float16 v16h;
typedef __attribute__((ext_vector_type(8)))  float    v8f;
typedef __attribute__((ext_vector_type(4)))  unsigned int u32x4;

union Frag { v16h h; u32x4 u[2]; };

// Problem dims (fixed by the reference)
constexpr int B_  = 2;
constexpr int N_  = 2048;
constexpr int C_  = 1024;
constexpr int H_  = 16;
constexpr int HD  = 64;
constexpr int M_  = B_ * N_;    // 4096 rows
constexpr int K3  = 3 * C_;     // 3072

// Workspace layout (bytes); total ~48 MB
constexpr size_t OFF_XH    = 0;
constexpr size_t OFF_WQKV  = OFF_XH   + (size_t)M_ * C_ * 2;
constexpr size_t OFF_WPROJ = OFF_WQKV + (size_t)K3 * C_ * 2;
constexpr size_t OFF_Q     = OFF_WPROJ + (size_t)C_ * C_ * 2;
constexpr size_t OFF_K     = OFF_Q + (size_t)B_ * H_ * N_ * HD * 2;
constexpr size_t OFF_V     = OFF_K + (size_t)B_ * H_ * N_ * HD * 2;
constexpr size_t OFF_AO    = OFF_V + (size_t)B_ * H_ * N_ * HD * 2;

// --------------------------- CDNA5 async global->LDS DMA (ASYNCcnt-tracked)
// Per-lane 16B copy, no VGPR staging. VDST holds the LDS byte address
// (low 32 bits of the generic pointer), VADDR the 64-bit global address.
__device__ __forceinline__ void async_ld_b128(const void* g, void* l) {
  asm volatile("global_load_async_to_lds_b128 %0, %1, off"
               :: "v"((unsigned)(uintptr_t)l),
                  "v"((unsigned long long)(uintptr_t)g)
               : "memory");
}
template <int N>
__device__ __forceinline__ void wait_async() {
  asm volatile("s_wait_asynccnt %0" :: "n"(N) : "memory");
}

// ---------------------------------------------------------------- convert
__global__ void cvt_f32_f16(const float* __restrict__ src,
                            _Float16* __restrict__ dst, int n) {
  for (int i = blockIdx.x * blockDim.x + threadIdx.x; i < n;
       i += gridDim.x * blockDim.x)
    dst[i] = (_Float16)src[i];
}

// ------------------------------------------------- QKV GEMM + bias + scatter
__global__ __launch_bounds__(256)
void qkv_gemm_scatter(const _Float16* __restrict__ A,
                      const _Float16* __restrict__ Wt,
                      const float* __restrict__ qb, const float* __restrict__ kb,
                      const float* __restrict__ vb,
                      _Float16* __restrict__ Qb, _Float16* __restrict__ Kb,
                      _Float16* __restrict__ Vb) {
  constexpr int BM = 128, BN = 128, BK = 32, LDA = 40, LDB = 40;
  constexpr int NS = C_ / BK;
  __shared__ _Float16 As[2][BM * LDA];
  __shared__ _Float16 Bs[2][BN * LDB];
  const int tid  = threadIdx.x;
  const int lane = tid & 31, wv = tid >> 5;
  const int wr = wv >> 2, wc = wv & 3;          // 2x4 wave grid, 64x32 per wave
  const int m0 = blockIdx.y * BM, n0 = blockIdx.x * BN;
  const int arow = lane & 15, ahalf = lane >> 4;

  v8f acc[4][2];
#pragma unroll
  for (int i = 0; i < 4; ++i)
#pragma unroll
    for (int j = 0; j < 2; ++j) acc[i][j] = {};

  // 4 async instructions per wave per stage (2 for A, 2 for B)
  auto issue = [&](int bf, int k0) {
#pragma unroll
    for (int it = 0; it < 2; ++it) {
      int i = tid + it * 256;
      int r = i >> 2, c = (i & 3) * 8;
      async_ld_b128(&A[(size_t)(m0 + r) * C_ + k0 + c], &As[bf][r * LDA + c]);
    }
#pragma unroll
    for (int it = 0; it < 2; ++it) {
      int i = tid + it * 256;
      int r = i >> 2, c = (i & 3) * 8;
      async_ld_b128(&Wt[(size_t)(n0 + r) * C_ + k0 + c], &Bs[bf][r * LDB + c]);
    }
  };

  issue(0, 0);
  for (int s = 0; s < NS; ++s) {
    const int cur = s & 1;
    if (s + 1 < NS) { issue(cur ^ 1, (s + 1) * BK); wait_async<4>(); }
    else            { wait_async<0>(); }
    __syncthreads();

    const _Float16* as = As[cur];
    const _Float16* bs = Bs[cur];
    Frag a[4], b[2];
#pragma unroll
    for (int mi = 0; mi < 4; ++mi) {
      int rb = (wr * 64 + mi * 16 + arow) * LDA + ahalf * 8;
      a[mi].u[0] = *(const u32x4*)&as[rb];
      a[mi].u[1] = *(const u32x4*)&as[rb + 16];
    }
#pragma unroll
    for (int ni = 0; ni < 2; ++ni) {
      int rb = (wc * 32 + ni * 16 + arow) * LDB + ahalf * 8;
      b[ni].u[0] = *(const u32x4*)&bs[rb];
      b[ni].u[1] = *(const u32x4*)&bs[rb + 16];
    }
#pragma unroll
    for (int mi = 0; mi < 4; ++mi)
#pragma unroll
      for (int ni = 0; ni < 2; ++ni)
        acc[mi][ni] = __builtin_amdgcn_wmma_f32_16x16x32_f16(
            false, a[mi].h, false, b[ni].h, (short)0, acc[mi][ni], false, false);
    __syncthreads();
  }

  // Epilogue: bias, fold softmax scale into Q, scatter to [B,H,N,hd] f16
  const float scale = 0.125f;  // hd^-0.5; reference's alpha trick is a no-op
#pragma unroll
  for (int mi = 0; mi < 4; ++mi)
#pragma unroll
    for (int ni = 0; ni < 2; ++ni)
#pragma unroll
      for (int r = 0; r < 8; ++r) {
        int m  = m0 + wr * 64 + mi * 16 + ahalf * 8 + r;
        int cN = n0 + wc * 32 + ni * 16 + arow;
        int bb = m >> 11, nn = m & (N_ - 1);
        int sec = cN >> 10, c2 = cN & (C_ - 1);
        int h = c2 >> 6, d = c2 & (HD - 1);
        float v = acc[mi][ni][r] + (sec == 0 ? qb : sec == 1 ? kb : vb)[c2];
        if (sec == 0) v *= scale;
        _Float16* dst = sec == 0 ? Qb : sec == 1 ? Kb : Vb;
        dst[(((size_t)bb * H_ + h) * N_ + nn) * HD + d] = (_Float16)v;
      }
}

// ------------------------------------------------------------ flash attention
__global__ __launch_bounds__(128)
void flash_attn(const _Float16* __restrict__ Qb, const _Float16* __restrict__ Kb,
                const _Float16* __restrict__ Vb, const float* __restrict__ mask,
                _Float16* __restrict__ Oh) {
  constexpr int BKV = 64, LDK = 72, LDV = 72, LDP = 72;
  constexpr int NKV = N_ / BKV;
  __shared__ _Float16 Ks[2][BKV * LDK];   // K tile [n][d], async DMA
  __shared__ _Float16 Vt[2][HD * LDV];    // V tile transposed [d][n]
  __shared__ _Float16 Pl[4 * 16 * LDP];   // per-wave P slabs

  const int tid = threadIdx.x, lane = tid & 31, wv = tid >> 5;
  const int bh = blockIdx.y, b = bh >> 4, h = bh & 15;
  const int qm0 = blockIdx.x * 64;
  const _Float16* Qp = Qb + (size_t)bh * N_ * HD;
  const _Float16* Kp = Kb + (size_t)bh * N_ * HD;
  const _Float16* Vp = Vb + (size_t)bh * N_ * HD;
  const float* mp = mask + (size_t)b * N_ * N_;
  const int arow = lane & 15, ahalf = lane >> 4;
  const int mrb = ahalf * 8;  // this lane's first local row in C-layout

  // Q fragments (A layout), loaded once: 16 rows x 64 d, 2 k-steps
  Frag qf[2];
  {
    const _Float16* qr = Qp + (size_t)(qm0 + wv * 16 + arow) * HD;
#pragma unroll
    for (int ks = 0; ks < 2; ++ks) {
      qf[ks].u[0] = *(const u32x4*)&qr[ks * 32 + ahalf * 8];
      qf[ks].u[1] = *(const u32x4*)&qr[ks * 32 + 16 + ahalf * 8];
    }
  }

  v8f o[4];
#pragma unroll
  for (int i = 0; i < 4; ++i) o[i] = {};
  float mrun[8], lsum[8];
#pragma unroll
  for (int r = 0; r < 8; ++r) { mrun[r] = -1e30f; lsum[r] = 0.f; }

  // 4 async instructions per wave per stage
  auto issueK = [&](int bf, int j0) {
#pragma unroll
    for (int it = 0; it < 4; ++it) {
      int i = tid + it * 128;
      int r = i >> 3, c = (i & 7) * 8;
      async_ld_b128(&Kp[(size_t)(j0 + r) * HD + c], &Ks[bf][r * LDK + c]);
    }
  };
  auto loadV = [&](int bf, int j0) {   // register-path transpose into LDS
#pragma unroll
    for (int it = 0; it < 4; ++it) {
      int i = tid + it * 128;
      int r = i >> 3, c = (i & 7) * 8;
      union { u32x4 u; _Float16 f[8]; } cv;
      cv.u = *(const u32x4*)&Vp[(size_t)(j0 + r) * HD + c];
#pragma unroll
      for (int e = 0; e < 8; ++e) Vt[bf][(c + e) * LDV + r] = cv.f[e];
    }
  };

  issueK(0, 0);
  loadV(0, 0);
  for (int s = 0; s < NKV; ++s) {
    const int cur = s & 1;
    const int j0 = s * BKV;
    if (s + 1 < NKV) {
      issueK(cur ^ 1, j0 + BKV);
      loadV(cur ^ 1, j0 + BKV);
      wait_async<4>();
    } else {
      wait_async<0>();
    }
    __syncthreads();

    const _Float16* ks_ = Ks[cur];
    const _Float16* vt_ = Vt[cur];

    // S = Q K^T over 4 n-tiles (f32 acc)
    v8f s_[4];
#pragma unroll
    for (int nt = 0; nt < 4; ++nt) {
      v8f sa = {};
#pragma unroll
      for (int ks = 0; ks < 2; ++ks) {
        Frag bk;
        int rb = (nt * 16 + arow) * LDK + ks * 32 + ahalf * 8;
        bk.u[0] = *(const u32x4*)&ks_[rb];
        bk.u[1] = *(const u32x4*)&ks_[rb + 16];
        sa = __builtin_amdgcn_wmma_f32_16x16x32_f16(
            false, qf[ks].h, false, bk.h, (short)0, sa, false, false);
      }
      s_[nt] = sa;
    }
    // additive mask
#pragma unroll
    for (int nt = 0; nt < 4; ++nt)
#pragma unroll
      for (int r = 0; r < 8; ++r) {
        int qrow = qm0 + wv * 16 + mrb + r;
        s_[nt][r] += mp[(size_t)qrow * N_ + j0 + nt * 16 + arow];
      }
    // online softmax: rows live in 16-lane halves -> xor-shuffle reduce
    float sf[8];
#pragma unroll
    for (int r = 0; r < 8; ++r) {
      float tm = s_[0][r];
#pragma unroll
      for (int nt = 1; nt < 4; ++nt) tm = fmaxf(tm, s_[nt][r]);
#pragma unroll
      for (int off = 1; off <= 8; off <<= 1)
        tm = fmaxf(tm, __shfl_xor(tm, off, 32));
      float nm = fmaxf(mrun[r], tm);
      sf[r] = __expf(mrun[r] - nm);
      mrun[r] = nm;
      float ps = 0.f;
#pragma unroll
      for (int nt = 0; nt < 4; ++nt) {
        float e = __expf(s_[nt][r] - nm);
        s_[nt][r] = e;
        ps += e;
      }
#pragma unroll
      for (int off = 1; off <= 8; off <<= 1) ps += __shfl_xor(ps, off, 32);
      lsum[r] = lsum[r] * sf[r] + ps;
#pragma unroll
      for (int dt = 0; dt < 4; ++dt) o[dt][r] *= sf[r];
    }
    // C-layout -> A-layout via wave-private LDS slab (in-order per-wave DS)
    _Float16* pw = &Pl[wv * 16 * LDP];
#pragma unroll
    for (int nt = 0; nt < 4; ++nt)
#pragma unroll
      for (int r = 0; r < 8; ++r)
        pw[(mrb + r) * LDP + nt * 16 + arow] = (_Float16)s_[nt][r];
    // O += P V
#pragma unroll
    for (int nc = 0; nc < 2; ++nc) {
      Frag pa;
      pa.u[0] = *(const u32x4*)&pw[arow * LDP + nc * 32 + ahalf * 8];
      pa.u[1] = *(const u32x4*)&pw[arow * LDP + nc * 32 + 16 + ahalf * 8];
#pragma unroll
      for (int dt = 0; dt < 4; ++dt) {
        Frag vb_;
        int rb = (dt * 16 + arow) * LDV + nc * 32 + ahalf * 8;
        vb_.u[0] = *(const u32x4*)&vt_[rb];
        vb_.u[1] = *(const u32x4*)&vt_[rb + 16];
        o[dt] = __builtin_amdgcn_wmma_f32_16x16x32_f16(
            false, pa.h, false, vb_.h, (short)0, o[dt], false, false);
      }
    }
    __syncthreads();
  }

  // normalize + store to [B, N, C] f16 (per-head column slice)
#pragma unroll
  for (int dt = 0; dt < 4; ++dt)
#pragma unroll
    for (int r = 0; r < 8; ++r) {
      int qrow = qm0 + wv * 16 + mrb + r;
      float v = o[dt][r] / lsum[r];
      Oh[((size_t)b * N_ + qrow) * C_ + h * HD + dt * 16 + arow] = (_Float16)v;
    }
}

// ------------------------------------------------------- output projection
__global__ __launch_bounds__(256)
void proj_gemm(const _Float16* __restrict__ A, const _Float16* __restrict__ Wt,
               const float* __restrict__ bp, float* __restrict__ out) {
  constexpr int BM = 128, BN = 128, BK = 32, LDA = 40, LDB = 40;
  constexpr int NS = C_ / BK;
  __shared__ _Float16 As[2][BM * LDA];
  __shared__ _Float16 Bs[2][BN * LDB];
  const int tid  = threadIdx.x;
  const int lane = tid & 31, wv = tid >> 5;
  const int wr = wv >> 2, wc = wv & 3;
  const int m0 = blockIdx.y * BM, n0 = blockIdx.x * BN;
  const int arow = lane & 15, ahalf = lane >> 4;

  v8f acc[4][2];
#pragma unroll
  for (int i = 0; i < 4; ++i)
#pragma unroll
    for (int j = 0; j < 2; ++j) acc[i][j] = {};

  auto issue = [&](int bf, int k0) {
#pragma unroll
    for (int it = 0; it < 2; ++it) {
      int i = tid + it * 256;
      int r = i >> 2, c = (i & 3) * 8;
      async_ld_b128(&A[(size_t)(m0 + r) * C_ + k0 + c], &As[bf][r * LDA + c]);
    }
#pragma unroll
    for (int it = 0; it < 2; ++it) {
      int i = tid + it * 256;
      int r = i >> 2, c = (i & 3) * 8;
      async_ld_b128(&Wt[(size_t)(n0 + r) * C_ + k0 + c], &Bs[bf][r * LDB + c]);
    }
  };

  issue(0, 0);
  for (int s = 0; s < NS; ++s) {
    const int cur = s & 1;
    if (s + 1 < NS) { issue(cur ^ 1, (s + 1) * BK); wait_async<4>(); }
    else            { wait_async<0>(); }
    __syncthreads();

    const _Float16* as = As[cur];
    const _Float16* bs = Bs[cur];
    Frag a[4], b[2];
#pragma unroll
    for (int mi = 0; mi < 4; ++mi) {
      int rb = (wr * 64 + mi * 16 + arow) * LDA + ahalf * 8;
      a[mi].u[0] = *(const u32x4*)&as[rb];
      a[mi].u[1] = *(const u32x4*)&as[rb + 16];
    }
#pragma unroll
    for (int ni = 0; ni < 2; ++ni) {
      int rb = (wc * 32 + ni * 16 + arow) * LDB + ahalf * 8;
      b[ni].u[0] = *(const u32x4*)&bs[rb];
      b[ni].u[1] = *(const u32x4*)&bs[rb + 16];
    }
#pragma unroll
    for (int mi = 0; mi < 4; ++mi)
#pragma unroll
      for (int ni = 0; ni < 2; ++ni)
        acc[mi][ni] = __builtin_amdgcn_wmma_f32_16x16x32_f16(
            false, a[mi].h, false, b[ni].h, (short)0, acc[mi][ni], false, false);
    __syncthreads();
  }
#pragma unroll
  for (int mi = 0; mi < 4; ++mi)
#pragma unroll
    for (int ni = 0; ni < 2; ++ni)
#pragma unroll
      for (int r = 0; r < 8; ++r) {
        int m  = m0 + wr * 64 + mi * 16 + ahalf * 8 + r;
        int cN = n0 + wc * 32 + ni * 16 + arow;
        out[(size_t)m * C_ + cN] = acc[mi][ni][r] + bp[cN];
      }
}

// --------------------------------------------------------------------- host
extern "C" void kernel_launch(void* const* d_in, const int* in_sizes, int n_in,
                              void* d_out, int out_size, void* d_ws,
                              size_t ws_size, hipStream_t stream) {
  (void)in_sizes; (void)n_in; (void)out_size; (void)ws_size;
  const float* x     = (const float*)d_in[0];
  const float* mask  = (const float*)d_in[1];
  const float* wqkv  = (const float*)d_in[2];
  const float* qb    = (const float*)d_in[3];
  const float* kb    = (const float*)d_in[4];
  const float* vb    = (const float*)d_in[5];
  const float* wproj = (const float*)d_in[6];
  const float* bproj = (const float*)d_in[7];
  float* out = (float*)d_out;

  char* ws = (char*)d_ws;
  _Float16* x_h     = (_Float16*)(ws + OFF_XH);
  _Float16* wqkv_h  = (_Float16*)(ws + OFF_WQKV);
  _Float16* wproj_h = (_Float16*)(ws + OFF_WPROJ);
  _Float16* Qb      = (_Float16*)(ws + OFF_Q);
  _Float16* Kb      = (_Float16*)(ws + OFF_K);
  _Float16* Vb      = (_Float16*)(ws + OFF_V);
  _Float16* attn_h  = (_Float16*)(ws + OFF_AO);

  cvt_f32_f16<<<2048, 256, 0, stream>>>(x, x_h, M_ * C_);
  cvt_f32_f16<<<2048, 256, 0, stream>>>(wqkv, wqkv_h, K3 * C_);
  cvt_f32_f16<<<1024, 256, 0, stream>>>(wproj, wproj_h, C_ * C_);

  qkv_gemm_scatter<<<dim3(K3 / 128, M_ / 128), 256, 0, stream>>>(
      x_h, wqkv_h, qb, kb, vb, Qb, Kb, Vb);

  flash_attn<<<dim3(N_ / 64, B_ * H_), 128, 0, stream>>>(
      Qb, Kb, Vb, mask, attn_h);

  proj_gemm<<<dim3(C_ / 128, M_ / 128), 256, 0, stream>>>(
      attn_h, wproj_h, bproj, out);
}
